// LiquidTimeConstant_16544214024864
// MI455X (gfx1250) — matrix operand order repaired and checked by reading, not compiled, not run on hardware
//
#include <hip/hip_runtime.h>
#include <hip/hip_bf16.h>

typedef __attribute__((ext_vector_type(16))) __bf16 v16bf;
typedef __attribute__((ext_vector_type(8)))  float  v8f;

#define UNITS   4096
#define TILE_K  32
#define LDA     40   // padded LDS row stride (bf16 elems); 80B rows keep b128 alignment
#define LDB     40

union Frag {
    uint4 q[2];
    v16bf v;
};

// pack high halves of two fp32 (bf16 truncation) into one dword: single v_perm_b32
__device__ __forceinline__ unsigned pk_hi16(float lo, float hi) {
    return __builtin_amdgcn_perm(__float_as_uint(hi), __float_as_uint(lo), 0x07060302u);
}

__global__ __launch_bounds__(256)
void ltc_wmma_kernel(const float* __restrict__ inputs,
                     const float* __restrict__ states,
                     const float* __restrict__ tau,
                     const float* __restrict__ W,
                     const float* __restrict__ sigma,
                     float* __restrict__ out)
{
    __shared__ unsigned short ldsA[2][256 * LDA];  // double-buffered [m][k] bf16
    __shared__ unsigned short ldsB[2][128 * LDB];  // double-buffered [n][k] bf16 (W^T)

    const int tid   = threadIdx.x;
    const int lane  = tid & 31;          // wave32
    const int wid   = tid >> 5;          // 8 waves
    const int waveM = wid >> 1;          // 0..3 -> 64 rows each
    const int waveN = wid & 1;           // 0..1 -> 64 cols each
    const int lmod  = lane & 15;
    const int lhal  = lane >> 4;

    const int rowBlock = blockIdx.y * 256;
    const int colBlock = blockIdx.x * 128;

    // --- staging assignments ---
    // A: one thread per row: 32 contiguous floats (8x b128 loads) -> 4x ds_store_b128
    const float* aSrc = inputs + (size_t)(rowBlock + tid) * UNITS;
    const int aDstIdx = tid * LDA;

    // B: thread -> (col, k-half): 16-row column gather of W (coalesced across lanes)
    const int b_n = tid & 127;
    const int b_h = tid >> 7;
    const float* bSrc = W + (size_t)(b_h * 16) * UNITS + (colBlock + b_n);
    const int bDstIdx = b_n * LDB + b_h * 16;

    v8f acc[4][4];
#pragma unroll
    for (int mt = 0; mt < 4; ++mt)
#pragma unroll
        for (int nt = 0; nt < 4; ++nt)
            acc[mt][nt] = 0.0f;

    float4 ta[8];
    float  tb[16];

    // ---- prologue: load K-tile 0 into registers ----
    {
        const float4* ap = (const float4*)aSrc;
#pragma unroll
        for (int i = 0; i < 8; ++i) ta[i] = ap[i];
#pragma unroll
        for (int i = 0; i < 16; ++i) tb[i] = bSrc[(size_t)i * UNITS];
    }
    // ---- pack + store tile 0 into buffer 0 ----
    {
        uint4* ad = (uint4*)&ldsA[0][aDstIdx];
#pragma unroll
        for (int i = 0; i < 4; ++i) {
            uint4 p;
            p.x = pk_hi16(ta[2 * i].x,     ta[2 * i].y);
            p.y = pk_hi16(ta[2 * i].z,     ta[2 * i].w);
            p.z = pk_hi16(ta[2 * i + 1].x, ta[2 * i + 1].y);
            p.w = pk_hi16(ta[2 * i + 1].z, ta[2 * i + 1].w);
            ad[i] = p;
        }
        uint4* bd = (uint4*)&ldsB[0][bDstIdx];
        uint4 p0, p1;
        p0.x = pk_hi16(tb[0],  tb[1]);  p0.y = pk_hi16(tb[2],  tb[3]);
        p0.z = pk_hi16(tb[4],  tb[5]);  p0.w = pk_hi16(tb[6],  tb[7]);
        p1.x = pk_hi16(tb[8],  tb[9]);  p1.y = pk_hi16(tb[10], tb[11]);
        p1.z = pk_hi16(tb[12], tb[13]); p1.w = pk_hi16(tb[14], tb[15]);
        bd[0] = p0;
        bd[1] = p1;
    }

    int buf = 0;
    for (int kt = 0; kt < UNITS; kt += TILE_K) {
        const bool hasNext = (kt + TILE_K) < UNITS;

        // ---- issue global loads for next K-tile (latency overlaps WMMA below) ----
        if (hasNext) {
            const float4* ap = (const float4*)(aSrc + kt + TILE_K);
#pragma unroll
            for (int i = 0; i < 8; ++i) ta[i] = ap[i];
            const float* bp = bSrc + (size_t)(kt + TILE_K) * UNITS;
#pragma unroll
            for (int i = 0; i < 16; ++i) tb[i] = bp[(size_t)i * UNITS];
            // prefetch tile k+2 into cache (global_prefetch_b8)
            if (kt + 2 * TILE_K < UNITS) {
                __builtin_prefetch(aSrc + kt + 2 * TILE_K, 0, 1);
                __builtin_prefetch(bSrc + (size_t)(kt + 2 * TILE_K) * UNITS, 0, 1);
            }
        }

        __syncthreads();   // buffer `buf` (written last iteration) now valid

        // ---- load fragments from LDS buffer `buf` (ds_load_b128 pairs) ----
        const unsigned short* bufA = ldsA[buf];
        const unsigned short* bufB = ldsB[buf];
        Frag afr[4], bfr[4];
#pragma unroll
        for (int mt = 0; mt < 4; ++mt) {
            const int m = waveM * 64 + mt * 16 + lmod;
            afr[mt].q[0] = *(const uint4*)&bufA[m * LDA + lhal * 16];
            afr[mt].q[1] = *(const uint4*)&bufA[m * LDA + lhal * 16 + 8];
        }
#pragma unroll
        for (int nt = 0; nt < 4; ++nt) {
            const int n = waveN * 64 + nt * 16 + lmod;
            bfr[nt].q[0] = *(const uint4*)&bufB[n * LDB + lhal * 16];
            bfr[nt].q[1] = *(const uint4*)&bufB[n * LDB + lhal * 16 + 8];
        }

        // ---- 16x v_wmma_f32_16x16x32_bf16 ----
#pragma unroll
        for (int mt = 0; mt < 4; ++mt)
#pragma unroll
            for (int nt = 0; nt < 4; ++nt)
                acc[mt][nt] = __builtin_amdgcn_wmma_f32_16x16x32_bf16(
                    false, afr[mt].v, false, bfr[nt].v,
                    (short)0, acc[mt][nt], false, false);

        // ---- pack + store next tile into the other buffer (disjoint from reads) ----
        if (hasNext) {
            uint4* ad = (uint4*)&ldsA[buf ^ 1][aDstIdx];
#pragma unroll
            for (int i = 0; i < 4; ++i) {
                uint4 p;
                p.x = pk_hi16(ta[2 * i].x,     ta[2 * i].y);
                p.y = pk_hi16(ta[2 * i].z,     ta[2 * i].w);
                p.z = pk_hi16(ta[2 * i + 1].x, ta[2 * i + 1].y);
                p.w = pk_hi16(ta[2 * i + 1].z, ta[2 * i + 1].w);
                ad[i] = p;
            }
            uint4* bd = (uint4*)&ldsB[buf ^ 1][bDstIdx];
            uint4 p0, p1;
            p0.x = pk_hi16(tb[0],  tb[1]);  p0.y = pk_hi16(tb[2],  tb[3]);
            p0.z = pk_hi16(tb[4],  tb[5]);  p0.w = pk_hi16(tb[6],  tb[7]);
            p1.x = pk_hi16(tb[8],  tb[9]);  p1.y = pk_hi16(tb[10], tb[11]);
            p1.z = pk_hi16(tb[12], tb[13]); p1.w = pk_hi16(tb[14], tb[15]);
            bd[0] = p0;
            bd[1] = p1;
        }

        buf ^= 1;
    }

    // ---- epilogue: sigmoid gate + exact ODE decay blend ----
#pragma unroll
    for (int nt = 0; nt < 4; ++nt) {
        const int col = colBlock + waveN * 64 + nt * 16 + lmod;
        const float sg  = sigma[col];
        const float dec = __expf(-0.1f / tau[col]);
#pragma unroll
        for (int mt = 0; mt < 4; ++mt) {
            const int rowBase = rowBlock + waveM * 64 + mt * 16 + lhal * 8;
#pragma unroll
            for (int r = 0; r < 8; ++r) {
                const size_t idx = (size_t)(rowBase + r) * UNITS + col;
                const float x = acc[mt][nt][r] + sg;
                const float a = 1.0f / (1.0f + __expf(-x));
                const float s = states[idx];
                out[idx] = a + (s - a) * dec;
            }
        }
    }
}

extern "C" void kernel_launch(void* const* d_in, const int* in_sizes, int n_in,
                              void* d_out, int out_size, void* d_ws, size_t ws_size,
                              hipStream_t stream) {
    const float* inputs = (const float*)d_in[0];
    const float* states = (const float*)d_in[1];
    const float* tau    = (const float*)d_in[2];
    const float* W      = (const float*)d_in[3];
    const float* sigma  = (const float*)d_in[4];
    float* out = (float*)d_out;

    const int units = in_sizes[2];            // 4096
    const int batch = in_sizes[0] / units;    // 8192

    dim3 grid(units / 128, batch / 256);      // 32 x 32 blocks
    dim3 block(256);                          // 8 wave32 waves
    hipLaunchKernelGGL(ltc_wmma_kernel, grid, block, 0, stream,
                       inputs, states, tau, W, sigma, out);
}
